// HGTGraphUpdate_66949950210771
// MI455X (gfx1250) — matrix operand order repaired and checked by reading, not compiled, not run on hardware
//
#include <hip/hip_runtime.h>
#include <math.h>

// ---------------------------------------------------------------------------
// HGT graph update for MI455X (gfx1250, wave32, WMMA).
//
// Strategy: fold per-head relation matrices (Watt/Wmsg, per edge TYPE) into the
// D x D node projections as block-diagonal composites, so all node-level math is
// one [N,128] x [128,640] mixed-precision WMMA GEMM producing
//   nodefeat[n] = [ q | ka0 | ka1 | ms0 | ms1 ]   (5 x 128 columns)
// The edge phase is then pure dot-products + segment softmax + weighted scatter,
// which is bandwidth/atomic bound (~2-3 GB traffic; hot arrays ~fit 192MB L2).
// Final gelu->GEMM also runs on WMMA; skip+LayerNorm is a wave-per-row pass.
// ---------------------------------------------------------------------------

#define NN 100000
#define DD 128
#define HH 8
#define CCH 16
#define EEDG 400000

typedef __attribute__((ext_vector_type(16))) _Float16 v16h;
typedef __attribute__((ext_vector_type(8)))  float    v8f;

// ---- workspace layout (bytes) ----
static constexpr size_t OFF_WBIG   = 0;                                    // f16 [128][640]
static constexpr size_t OFF_BBIG   = OFF_WBIG + (size_t)128 * 640 * 2;     // f32 [640]
static constexpr size_t OFF_WAH    = OFF_BBIG + (size_t)640 * 4;           // f16 [128][128]
static constexpr size_t OFF_NF     = ((OFF_WAH + (size_t)128 * 128 * 2 + 511) / 512) * 512; // f32 [N][640]
static constexpr size_t OFF_SCORES = OFF_NF     + (size_t)NN * 640 * 4;    // f32 [2E][8]
static constexpr size_t OFF_SMAX   = OFF_SCORES + (size_t)2 * EEDG * 8 * 4;// f32 [N][8]
static constexpr size_t OFF_DENOM  = OFF_SMAX   + (size_t)NN * 8 * 4;      // f32 [N][8]
static constexpr size_t OFF_POOLED = OFF_DENOM  + (size_t)NN * 8 * 4;      // f32 [N][128]
static constexpr size_t OFF_HBUF   = OFF_POOLED + (size_t)NN * 128 * 4;    // f32 [N][128]

__device__ __forceinline__ void atomicMaxF(float* addr, float val) {
  // sign-aware int trick -> lowers to native global_atomic_max/min (no CAS loop)
  if (val >= 0.0f) atomicMax((int*)addr, __float_as_int(val));
  else             atomicMin((unsigned int*)addr, (unsigned int)__float_as_int(val));
}

__device__ __forceinline__ float gelu_exact(float v) {
  return 0.5f * v * (1.0f + erff(v * 0.70710678118654752f));
}

// ---------------------------------------------------------------------------
// Kernel 1: build fused f16 weights.
//   cols   0..127 : Wq                      (bias bq)
//   cols 128..255 : Wk @ blockdiag(Watt0)   (bias bk @ BD(Watt0))
//   cols 256..383 : Wk @ blockdiag(Watt1)
//   cols 384..511 : Wm @ blockdiag(Wmsg0)
//   cols 512..639 : Wm @ blockdiag(Wmsg1)
// blocks 640..767 convert Wa -> f16.
// ---------------------------------------------------------------------------
__global__ void prep_weights(const float* Wk, const float* bk,
                             const float* Wm, const float* bm,
                             const float* Wq, const float* bq,
                             const float* Wa,
                             const float* Watt0, const float* Wmsg0,
                             const float* Watt1, const float* Wmsg1,
                             _Float16* WbigH, float* bbig, _Float16* WaH) {
  const int j  = blockIdx.x;   // output column (0..767)
  const int kk = threadIdx.x;  // input row     (0..127)
  if (j < 640) {
    float wv, bv = 0.0f;
    if (j < 128) {
      wv = Wq[kk * 128 + j];
      bv = bq[j];
    } else {
      const int grp = (j - 128) >> 7;  // 0:ka0 1:ka1 2:ms0 3:ms1
      const int jj  = (j - 128) & 127;
      const int h = jj >> 4, l = jj & 15;
      const float* Wnode = (grp < 2) ? Wk : Wm;
      const float* bnode = (grp < 2) ? bk : bm;
      const float* Wrel  = (grp == 0) ? Watt0 : (grp == 1) ? Watt1
                         : (grp == 2) ? Wmsg0 : Wmsg1;
      float s = 0.0f, sb = 0.0f;
      #pragma unroll
      for (int c = 0; c < 16; ++c) {
        const float r = Wrel[h * 256 + c * 16 + l];
        s  += Wnode[kk * 128 + h * 16 + c] * r;
        sb += bnode[h * 16 + c] * r;
      }
      wv = s; bv = sb;
    }
    WbigH[(size_t)kk * 640 + j] = (_Float16)wv;
    if (kk == 0) bbig[j] = bv;
  } else {
    const int jj = j - 640;
    WaH[(size_t)kk * 128 + jj] = (_Float16)Wa[(size_t)kk * 128 + jj];
  }
}

// ---------------------------------------------------------------------------
// Kernel 2: init segment buffers.
// ---------------------------------------------------------------------------
__global__ void init_buffers(float* smax, float* denom, float* pooled) {
  const int i = blockIdx.x * blockDim.x + threadIdx.x;
  const int stride = gridDim.x * blockDim.x;
  for (int t = i; t < NN * HH; t += stride) { smax[t] = -INFINITY; denom[t] = 0.0f; }
  for (int t = i; t < NN * DD; t += stride) { pooled[t] = 0.0f; }
}

// ---------------------------------------------------------------------------
// Kernel 3: fused node GEMM  nodefeat[N][640] = f16(x) @ Wbig + bbig.
// Block = 128 threads = 4 waves; each wave one 16x16 D-tile; block covers
// 16 rows x 64 cols. K=128 -> 4 x v_wmma_f32_16x16x32_f16.
// ---------------------------------------------------------------------------
__global__ __launch_bounds__(128)
void gemm_nodes(const float* __restrict__ x, const _Float16* __restrict__ WbigH,
                const float* __restrict__ bbig, float* __restrict__ nodefeat) {
  __shared__ _Float16 Ah[16][128];   // A tile, row-major, f16
  __shared__ _Float16 Bt[64][128];   // B tile transposed: Bt[col][k]

  const int tid  = threadIdx.x;
  const int lane = tid & 31;
  const int wid  = tid >> 5;
  const int rowBase = blockIdx.x * 16;
  const int colBase = blockIdx.y * 64;

  for (int idx = tid; idx < 16 * 128; idx += 128) {
    const int r = idx >> 7, c = idx & 127;
    Ah[r][c] = (_Float16)x[(size_t)(rowBase + r) * 128 + c];
  }
  for (int idx = tid; idx < 64 * 128; idx += 128) {
    const int c = idx >> 7, k = idx & 127;
    Bt[c][k] = WbigH[(size_t)k * 640 + colBase + c];
  }
  __syncthreads();

  const int mlo = lane & 15;   // M row (A) / N col (B,C,D)
  const int hi  = lane >> 4;   // lane-half selector
  const int col = wid * 16 + mlo;

  v8f acc = {};
  #pragma unroll
  for (int ks = 0; ks < 4; ++ks) {
    v16h a, b;
    // A 16x32 f16 layout: lanes 0-15 hold K 0-7 (v0-3) & 16-23 (v4-7); lanes 16-31 +8.
    const _Float16* Ar = &Ah[mlo][ks * 32 + hi * 8];
    #pragma unroll
    for (int i = 0; i < 4; ++i) {
      a[2 * i]     = Ar[2 * i];      a[2 * i + 1]     = Ar[2 * i + 1];
      a[8 + 2 * i] = Ar[16 + 2 * i]; a[8 + 2 * i + 1] = Ar[16 + 2 * i + 1];
    }
    // B 32x16 f16 layout: lanes 0-15 hold K 0-7 & 16-23; lanes 16-31 +8 (matches A).
    const _Float16* Br = &Bt[col][ks * 32 + hi * 8];
    #pragma unroll
    for (int i = 0; i < 4; ++i) {
      b[2 * i]     = Br[2 * i];      b[2 * i + 1]     = Br[2 * i + 1];
      b[8 + 2 * i] = Br[16 + 2 * i]; b[8 + 2 * i + 1] = Br[16 + 2 * i + 1];
    }
    acc = __builtin_amdgcn_wmma_f32_16x16x32_f16(false, a, false, b, (short)0, acc,
                                                 false, false);
  }

  const int gcol = colBase + col;
  const float bias = bbig[gcol];
  #pragma unroll
  for (int r = 0; r < 8; ++r) {
    const int m = rowBase + r + hi * 8;   // C/D layout: VGPR r -> M=r (+8 for hi lanes)
    nodefeat[(size_t)m * 640 + gcol] = acc[r] + bias;
  }
}

// ---------------------------------------------------------------------------
// Kernel 4: per-edge scores + segment max. Wave per edge: 8 heads x 4 lanes.
// ---------------------------------------------------------------------------
__global__ __launch_bounds__(256)
void edge_score(const int* __restrict__ src0, const int* __restrict__ dst0,
                const int* __restrict__ src1, const int* __restrict__ dst1,
                const float* __restrict__ prior0, const float* __restrict__ prior1,
                const float* __restrict__ nodefeat,
                float* __restrict__ scores, float* __restrict__ smax) {
  const int lane = threadIdx.x & 31;
  const int gw = blockIdx.x * (blockDim.x >> 5) + (threadIdx.x >> 5);
  const int nw = gridDim.x * (blockDim.x >> 5);
  const int h = lane >> 2;
  for (int e = gw; e < 2 * EEDG; e += nw) {
    const int eset = (e >= EEDG);
    const int ei = e - eset * EEDG;
    const int s_ = eset ? src1[ei] : src0[ei];
    const int d_ = eset ? dst1[ei] : dst0[ei];
    const float4 av = ((const float4*)(nodefeat + (size_t)s_ * 640 + 128 + eset * 128))[lane];
    const float4 bv = ((const float4*)(nodefeat + (size_t)d_ * 640))[lane];
    float p = av.x * bv.x + av.y * bv.y + av.z * bv.z + av.w * bv.w;
    p += __shfl_xor(p, 1, 32);
    p += __shfl_xor(p, 2, 32);           // lanes 4h..4h+3 hold head-h dot
    if ((lane & 3) == 0) {
      const float pr = eset ? prior1[h] : prior0[h];
      const float sc = p * pr * 0.25f;   // scale = 1/sqrt(C)
      scores[(size_t)e * 8 + h] = sc;
      atomicMaxF(&smax[(size_t)d_ * 8 + h], sc);
    }
  }
}

// ---------------------------------------------------------------------------
// Kernel 5: exp + segment sum (flat over 2E*H).
// ---------------------------------------------------------------------------
__global__ void edge_expsum(const int* __restrict__ dst0, const int* __restrict__ dst1,
                            const float* __restrict__ scores,
                            const float* __restrict__ smax, float* __restrict__ denom) {
  const int i = blockIdx.x * blockDim.x + threadIdx.x;
  const int stride = gridDim.x * blockDim.x;
  for (int t = i; t < 2 * EEDG * HH; t += stride) {
    const int e = t >> 3, h = t & 7;
    const int eset = (e >= EEDG);
    const int ei = e - eset * EEDG;
    const int d_ = eset ? dst1[ei] : dst0[ei];
    const float ex = expf(scores[t] - smax[(size_t)d_ * 8 + h]);
    atomicAdd(&denom[(size_t)d_ * 8 + h], ex);
  }
}

// ---------------------------------------------------------------------------
// Kernel 6: attention-weighted scatter of precomputed messages.
// ---------------------------------------------------------------------------
__global__ __launch_bounds__(256)
void edge_accum(const int* __restrict__ src0, const int* __restrict__ dst0,
                const int* __restrict__ src1, const int* __restrict__ dst1,
                const float* __restrict__ nodefeat, const float* __restrict__ scores,
                const float* __restrict__ smax, const float* __restrict__ denom,
                float* __restrict__ pooled) {
  const int lane = threadIdx.x & 31;
  const int gw = blockIdx.x * (blockDim.x >> 5) + (threadIdx.x >> 5);
  const int nw = gridDim.x * (blockDim.x >> 5);
  const int h = lane >> 2;
  for (int e = gw; e < 2 * EEDG; e += nw) {
    const int eset = (e >= EEDG);
    const int ei = e - eset * EEDG;
    const int s_ = eset ? src1[ei] : src0[ei];
    const int d_ = eset ? dst1[ei] : dst0[ei];
    const float sc = scores[(size_t)e * 8 + h];
    const float mx = smax[(size_t)d_ * 8 + h];
    const float dn = denom[(size_t)d_ * 8 + h];
    const float w = expf(sc - mx) / (dn > 0.0f ? dn : 1.0f);
    const float4 mv = ((const float4*)(nodefeat + (size_t)s_ * 640 + 384 + eset * 128))[lane];
    float* pp = pooled + (size_t)d_ * 128 + lane * 4;
    atomicAdd(pp + 0, w * mv.x);
    atomicAdd(pp + 1, w * mv.y);
    atomicAdd(pp + 2, w * mv.z);
    atomicAdd(pp + 3, w * mv.w);
  }
}

// ---------------------------------------------------------------------------
// Kernel 7: hbuf = gelu(pooled) @ Wa + ba  (WMMA, same tiling as gemm_nodes).
// ---------------------------------------------------------------------------
__global__ __launch_bounds__(128)
void gemm_final(const float* __restrict__ pooled, const _Float16* __restrict__ WaH,
                const float* __restrict__ ba, float* __restrict__ hbuf) {
  __shared__ _Float16 Ah[16][128];
  __shared__ _Float16 Bt[64][128];

  const int tid  = threadIdx.x;
  const int lane = tid & 31;
  const int wid  = tid >> 5;
  const int rowBase = blockIdx.x * 16;
  const int colBase = blockIdx.y * 64;

  for (int idx = tid; idx < 16 * 128; idx += 128) {
    const int r = idx >> 7, c = idx & 127;
    Ah[r][c] = (_Float16)gelu_exact(pooled[(size_t)(rowBase + r) * 128 + c]);
  }
  for (int idx = tid; idx < 64 * 128; idx += 128) {
    const int c = idx >> 7, k = idx & 127;
    Bt[c][k] = WaH[(size_t)k * 128 + colBase + c];
  }
  __syncthreads();

  const int mlo = lane & 15;
  const int hi  = lane >> 4;
  const int col = wid * 16 + mlo;

  v8f acc = {};
  #pragma unroll
  for (int ks = 0; ks < 4; ++ks) {
    v16h a, b;
    const _Float16* Ar = &Ah[mlo][ks * 32 + hi * 8];
    #pragma unroll
    for (int i = 0; i < 4; ++i) {
      a[2 * i]     = Ar[2 * i];      a[2 * i + 1]     = Ar[2 * i + 1];
      a[8 + 2 * i] = Ar[16 + 2 * i]; a[8 + 2 * i + 1] = Ar[16 + 2 * i + 1];
    }
    const _Float16* Br = &Bt[col][ks * 32 + hi * 8];
    #pragma unroll
    for (int i = 0; i < 4; ++i) {
      b[2 * i]     = Br[2 * i];      b[2 * i + 1]     = Br[2 * i + 1];
      b[8 + 2 * i] = Br[16 + 2 * i]; b[8 + 2 * i + 1] = Br[16 + 2 * i + 1];
    }
    acc = __builtin_amdgcn_wmma_f32_16x16x32_f16(false, a, false, b, (short)0, acc,
                                                 false, false);
  }

  const int gcol = colBase + col;
  const float bias = ba[gcol];
  #pragma unroll
  for (int r = 0; r < 8; ++r) {
    const int m = rowBase + r + hi * 8;
    hbuf[(size_t)m * 128 + gcol] = acc[r] + bias;
  }
}

// ---------------------------------------------------------------------------
// Kernel 8: out = LN( alpha*h + (1-alpha)*x ) * gamma + beta. Wave per row.
// ---------------------------------------------------------------------------
__global__ __launch_bounds__(256)
void finalize_ln(const float* __restrict__ hbuf, const float* __restrict__ x,
                 const float* __restrict__ skip, const float* __restrict__ gamma,
                 const float* __restrict__ beta, float* __restrict__ out) {
  const int lane = threadIdx.x & 31;
  const int row = blockIdx.x * (blockDim.x >> 5) + (threadIdx.x >> 5);
  if (row >= NN) return;
  const float alpha = 1.0f / (1.0f + expf(-skip[0]));
  const float4 hv = ((const float4*)(hbuf + (size_t)row * 128))[lane];
  const float4 xv = ((const float4*)(x    + (size_t)row * 128))[lane];
  float4 o;
  o.x = alpha * hv.x + (1.0f - alpha) * xv.x;
  o.y = alpha * hv.y + (1.0f - alpha) * xv.y;
  o.z = alpha * hv.z + (1.0f - alpha) * xv.z;
  o.w = alpha * hv.w + (1.0f - alpha) * xv.w;
  float s  = o.x + o.y + o.z + o.w;
  float s2 = o.x * o.x + o.y * o.y + o.z * o.z + o.w * o.w;
  #pragma unroll
  for (int m = 1; m < 32; m <<= 1) {
    s  += __shfl_xor(s,  m, 32);
    s2 += __shfl_xor(s2, m, 32);
  }
  const float mean = s * (1.0f / 128.0f);
  const float var  = s2 * (1.0f / 128.0f) - mean * mean;
  const float inv  = rsqrtf(var + 1e-3f);
  const float4 gv = ((const float4*)gamma)[lane];
  const float4 bv = ((const float4*)beta)[lane];
  float4 r;
  r.x = (o.x - mean) * inv * gv.x + bv.x;
  r.y = (o.y - mean) * inv * gv.y + bv.y;
  r.z = (o.z - mean) * inv * gv.z + bv.z;
  r.w = (o.w - mean) * inv * gv.w + bv.w;
  ((float4*)(out + (size_t)row * 128))[lane] = r;
}

// ---------------------------------------------------------------------------
extern "C" void kernel_launch(void* const* d_in, const int* in_sizes, int n_in,
                              void* d_out, int out_size, void* d_ws, size_t ws_size,
                              hipStream_t stream) {
  (void)in_sizes; (void)n_in; (void)out_size; (void)ws_size;
  const float* x     = (const float*)d_in[0];
  const int*   src0  = (const int*)d_in[1];
  const int*   dst0  = (const int*)d_in[2];
  const int*   src1  = (const int*)d_in[3];
  const int*   dst1  = (const int*)d_in[4];
  const float* Wk    = (const float*)d_in[5];
  const float* bk    = (const float*)d_in[6];
  const float* Wm    = (const float*)d_in[7];
  const float* bm    = (const float*)d_in[8];
  const float* Wq    = (const float*)d_in[9];
  const float* bq    = (const float*)d_in[10];
  const float* Wa    = (const float*)d_in[11];
  const float* ba    = (const float*)d_in[12];
  const float* Watt0 = (const float*)d_in[13];
  const float* Wmsg0 = (const float*)d_in[14];
  const float* Watt1 = (const float*)d_in[15];
  const float* Wmsg1 = (const float*)d_in[16];
  const float* prior0 = (const float*)d_in[17];
  const float* prior1 = (const float*)d_in[18];
  const float* skip   = (const float*)d_in[19];
  const float* gamma  = (const float*)d_in[20];
  const float* beta   = (const float*)d_in[21];

  char* ws = (char*)d_ws;
  _Float16* WbigH   = (_Float16*)(ws + OFF_WBIG);
  float*    bbig    = (float*)(ws + OFF_BBIG);
  _Float16* WaH     = (_Float16*)(ws + OFF_WAH);
  float*    nodefeat= (float*)(ws + OFF_NF);
  float*    scores  = (float*)(ws + OFF_SCORES);
  float*    smax    = (float*)(ws + OFF_SMAX);
  float*    denom   = (float*)(ws + OFF_DENOM);
  float*    pooled  = (float*)(ws + OFF_POOLED);
  float*    hbuf    = (float*)(ws + OFF_HBUF);

  prep_weights<<<768, 128, 0, stream>>>(Wk, bk, Wm, bm, Wq, bq, Wa,
                                        Watt0, Wmsg0, Watt1, Wmsg1,
                                        WbigH, bbig, WaH);
  init_buffers<<<2048, 256, 0, stream>>>(smax, denom, pooled);
  gemm_nodes<<<dim3(NN / 16, 10), 128, 0, stream>>>(x, WbigH, bbig, nodefeat);
  edge_score<<<4096, 256, 0, stream>>>(src0, dst0, src1, dst1, prior0, prior1,
                                       nodefeat, scores, smax);
  edge_expsum<<<8192, 256, 0, stream>>>(dst0, dst1, scores, smax, denom);
  edge_accum<<<4096, 256, 0, stream>>>(src0, dst0, src1, dst1, nodefeat, scores,
                                       smax, denom, pooled);
  gemm_final<<<dim3(NN / 16, 2), 128, 0, stream>>>(pooled, WaH, ba, hbuf);
  finalize_ln<<<12500, 256, 0, stream>>>(hbuf, x, skip, gamma, beta, (float*)d_out);
}
